// Output_cost3_41678362640718
// MI455X (gfx1250) — matrix operand order repaired and checked by reading, not compiled, not run on hardware
//
#include <hip/hip_runtime.h>

typedef __attribute__((ext_vector_type(16))) _Float16 v16h;
typedef __attribute__((ext_vector_type(8)))  float    v8f;

#define BB 2
#define CC 48     // image rows == bilateral "channels"
#define WW 128    // spatial dim 1 (w)
#define KK 128    // spatial dim 2 (k / disparity)

// ---- gfx1250 async global->LDS path (guarded; falls back to sync staging) ----
#if defined(__has_builtin)
#if __has_builtin(__builtin_amdgcn_global_load_async_to_lds_b32) && \
    __has_builtin(__builtin_amdgcn_s_wait_asynccnt)
#define USE_ASYNC_LDS 1
#endif
#endif

typedef __attribute__((address_space(1))) int g_int;
typedef __attribute__((address_space(3))) int l_int;

__device__ __forceinline__ void lds_fill_async(const float* g, float* l)
{
#ifdef USE_ASYNC_LDS
    __builtin_amdgcn_global_load_async_to_lds_b32(
        (g_int*)g, (l_int*)l, /*offset=*/0, /*cpol=*/0);
#else
    *l = *g;
#endif
}

__device__ __forceinline__ void lds_fill_wait()
{
#ifdef USE_ASYNC_LDS
    __builtin_amdgcn_s_wait_asynccnt(0);
#endif
}

__device__ __forceinline__ float sigmoidf(float x) { return 1.0f / (1.0f + __expf(-x)); }

// ---------------------------------------------------------------------------
// Kernel 1: banded softmax over k for both volumes.
// grid.x = B*C*W rows, grid.y = volume (0: r2l tril band, 1: l2r triu band)
// ---------------------------------------------------------------------------
__global__ void band_softmax_kernel(const float* __restrict__ cost_r2l,
                                    const float* __restrict__ cost_l2r,
                                    const int* __restrict__ mdp,
                                    float* __restrict__ S0,
                                    float* __restrict__ S1)
{
    const int row = blockIdx.x;          // b*C*W + c*W + w
    const int vol = blockIdx.y;
    const int t   = threadIdx.x;         // k index, 0..127
    const int w   = row % WW;
    const int md  = mdp[0];

    const float* src = vol ? cost_l2r : cost_r2l;
    float*       dst = vol ? S1 : S0;

    const bool m = vol ? (t >= w && t < w + md)
                       : (t <= w && t > w - md);
    const float x = src[(size_t)row * KK + t];
    const float banded = m ? x : 0.0f;   // reference takes rowmax including zeros

    __shared__ float red[128];
    red[t] = banded; __syncthreads();
    for (int s = 64; s > 0; s >>= 1) {
        if (t < s) red[t] = fmaxf(red[t], red[t + s]);
        __syncthreads();
    }
    const float mx = red[0]; __syncthreads();

    const float e = m ? __expf(banded - mx) : 0.0f;
    red[t] = e; __syncthreads();
    for (int s = 64; s > 0; s >>= 1) {
        if (t < s) red[t] += red[t + s];
        __syncthreads();
    }
    dst[(size_t)row * KK + t] = e / (red[0] + 1e-8f);
}

// ---------------------------------------------------------------------------
// Kernel 2: guidance = mean over channels (image rows).
// ---------------------------------------------------------------------------
__global__ void cmean_kernel(const float* __restrict__ S0,
                             const float* __restrict__ S1,
                             float* __restrict__ cm0,
                             float* __restrict__ cm1)
{
    const int wk  = blockIdx.x * blockDim.x + threadIdx.x;   // 0..W*K-1
    const int vol = blockIdx.y;
    const int b   = blockIdx.z;
    const float* S  = vol ? S1 : S0;
    float*       cm = vol ? cm1 : cm0;
    const float* p = S + (size_t)b * CC * WW * KK + wk;
    float s = 0.0f;
#pragma unroll 4
    for (int c = 0; c < CC; ++c) s += p[(size_t)c * WW * KK];
    cm[(size_t)b * WW * KK + wk] = s * (1.0f / (float)CC);
}

// ---------------------------------------------------------------------------
// Kernel 3: joint bilateral 5x5 + fused reduction.
//   DIR==0 (r2l): outsum[b,c,w] += sum_k filt[w,k]*k   via WMMA matvec
//   DIR==1 (l2r): outsum[b,c,k] += sum_w filt[w,k]     via f32 atomics
// One block = 16x16 (w,k) tile; threadIdx = (tx=k, ty=w).
// Halo tiles land via GLOBAL_LOAD_ASYNC_TO_LDS (ASYNCcnt) when available.
// ---------------------------------------------------------------------------
template <int DIR>
__global__ void bilateral_kernel(const float* __restrict__ S,
                                 const float* __restrict__ cmean,
                                 const float* __restrict__ params,
                                 float* __restrict__ outsum)
{
    __shared__ float    cmp[20 * 20];
    __shared__ float    xp[20 * 20];
    __shared__ _Float16 tileF[16][16];

    const int tx = threadIdx.x, ty = threadIdx.y;
    const int tid = ty * 16 + tx;
    const int b  = blockIdx.z;
    const int w0 = blockIdx.x * 16, k0 = blockIdx.y * 16;
    const int k  = k0 + tx;

    const float p0 = params[b * 3 + 0];
    const float s1 = 3.5f + 5.5f * sigmoidf(params[b * 3 + 1]);
    const float s2 = 5.5f + 7.5f * sigmoidf(params[b * 3 + 2]);
    const bool  is5 = ((int)p0) >= 0;                 // trunc-toward-zero semantics
    const float inv2s1 = 1.0f / (2.0f * s1 * s1);
    const float inv2s2 = 1.0f / (2.0f * s2 * s2);

    // guidance patch (zero-padded 20x20), async fill where in range
    const float* cmb = cmean + (size_t)b * WW * KK;
    for (int i = tid; i < 400; i += 256) {
        const int pw = w0 - 2 + i / 20, pk = k0 - 2 + i % 20;
        if (pw >= 0 && pw < WW && pk >= 0 && pk < KK)
            lds_fill_async(&cmb[pw * KK + pk], &cmp[i]);
        else
            cmp[i] = 0.0f;
    }
    lds_fill_wait();
    __syncthreads();

    const float c0 = cmp[(ty + 2) * 20 + (tx + 2)];

    // pass 1: spatial & color kernel sums
    float sksum = 0.0f, cksum = 0.0f;
#pragma unroll
    for (int d = 0; d < 5; ++d)
#pragma unroll
        for (int e = 0; e < 5; ++e) {
            const float km = is5 ? 1.0f
                                 : ((d >= 1 && d <= 3 && e >= 1 && e <= 3) ? 1.0f : 0.0f);
            const float gy = (float)d - 2.0f, gx = (float)e - 2.0f;
            sksum += __expf(-(gy * gy + gx * gx) * inv2s1) * km;
            const float diff = cmp[(ty + d) * 20 + (tx + e)] - c0;
            cksum += __expf(-diff * diff * inv2s2) * km;
        }
    const float iss = 1.0f / sksum, isc = 1.0f / cksum;

    // pass 2: fused normalized weights
    float wgt[25];
    float wsum = 0.0f;
#pragma unroll
    for (int d = 0; d < 5; ++d)
#pragma unroll
        for (int e = 0; e < 5; ++e) {
            const float km = is5 ? 1.0f
                                 : ((d >= 1 && d <= 3 && e >= 1 && e <= 3) ? 1.0f : 0.0f);
            const float gy = (float)d - 2.0f, gx = (float)e - 2.0f;
            const float sk = __expf(-(gy * gy + gx * gx) * inv2s1) * km * iss;
            const float diff = cmp[(ty + d) * 20 + (tx + e)] - c0;
            const float ck = __expf(-diff * diff * inv2s2) * km * isc;
            const float ww = sk * ck;
            wgt[d * 5 + e] = ww;
            wsum += ww;
        }
    const float iw = 1.0f / (wsum + 1e-8f);
#pragma unroll
    for (int i = 0; i < 25; ++i) wgt[i] *= iw;

    // channel loop: async tiled halo load -> 25-tap FMA -> fused reduction
    const float* Sb = S + (size_t)b * CC * WW * KK;
    for (int c = 0; c < CC; ++c) {
        __syncthreads();
        const float* Sc = Sb + (size_t)c * WW * KK;
        if (c + 1 < CC)   // gfx1250 global_prefetch_b8 for next channel's tile
            __builtin_prefetch(Sc + (size_t)WW * KK + (size_t)(w0 + ty) * KK + k, 0, 1);
        for (int i = tid; i < 400; i += 256) {
            const int pw = w0 - 2 + i / 20, pk = k0 - 2 + i % 20;
            if (pw >= 0 && pw < WW && pk >= 0 && pk < KK)
                lds_fill_async(&Sc[pw * KK + pk], &xp[i]);
            else
                xp[i] = 0.0f;
        }
        lds_fill_wait();
        __syncthreads();

        float f = 0.0f;
#pragma unroll
        for (int d = 0; d < 5; ++d)
#pragma unroll
            for (int e = 0; e < 5; ++e)
                f = fmaf(xp[(ty + d) * 20 + (tx + e)], wgt[d * 5 + e], f);

        if (DIR == 0) {
            // out[w] += sum_k filt[w,k] * k : WMMA matvec with broadcast B columns
            tileF[ty][tx] = (_Float16)f;
            __syncthreads();
            if (tid < 32) {                    // wave 0 only, EXEC all ones
                const int lane = tid;
                const int M = lane & 15;
                v16h a, bv;
#pragma unroll
                for (int h = 0; h < 16; ++h) {
                    // A (16x32 f16 layout): lanes 0-15 halves 0-7 = K 0..7,
                    // lanes 16-31 halves 0-7 = K 8..15; K 16..31 padded with 0.
                    _Float16 av = (_Float16)0.0f;
                    if (h < 8) av = tileF[M][(lane < 16) ? h : (8 + h)];
                    a[h] = av;
                    // B (32x16 f16): lanes 0-15 halves 0-7 = K 0..7, lanes 16-31 halves 0-7 = K 8..15; all columns = idx[k]
                    bv[h] = (h < 8) ? (_Float16)(float)(k0 + ((lane < 16) ? h : (8 + h))) : (_Float16)0.0f;
                }
                v8f acc = {};
                acc = __builtin_amdgcn_wmma_f32_16x16x32_f16(
                    false, a, false, bv, (short)0, acc, false, false);
                float* dsum = outsum + (size_t)b * CC * WW + (size_t)c * WW + w0;
                if (lane == 0) {
#pragma unroll
                    for (int r = 0; r < 8; ++r) atomicAdd(&dsum[r], acc[r]);
                }
                if (lane == 16) {
#pragma unroll
                    for (int r = 0; r < 8; ++r) atomicAdd(&dsum[8 + r], acc[r]);
                }
            }
        } else {
            // out[k] += sum_w filt[w,k]
            atomicAdd(&outsum[(size_t)b * CC * KK + (size_t)c * KK + k], f);
        }
    }
}

// ---------------------------------------------------------------------------
// Kernel 4: disparity regression + right/left iterative fill (one block/row).
// ---------------------------------------------------------------------------
__global__ void regress_fill_kernel(const float* __restrict__ dispsum,
                                    const float* __restrict__ colsum,
                                    float* __restrict__ out)
{
    const int row = blockIdx.x;   // b*C + c
    const int t   = threadIdx.x;  // 0..127
    __shared__ float d[WW], v[WW];

    const float di  = (float)t - dispsum[(size_t)row * WW + t];
    const float vmf = (colsum[(size_t)row * WW + t] > 0.1f) ? 1.0f : 0.0f;
    d[t] = di * vmf;
    v[t] = vmf;
    __syncthreads();

    for (int it = 0; it < WW; ++it) {          // rightward fill
        const float dl = (t > 0) ? d[t - 1] : 0.0f;
        const float vl = (t > 0) ? v[t - 1] : 0.0f;
        const float dc = d[t], v0 = v[t];
        __syncthreads();
        const float v1  = v0 + vl;
        const float v1p = (v1 > 0.0f) ? 1.0f : 0.0f;
        d[t] = dc * v0 + (dc + dl) / (v1 + 1e-4f) * (v1p - v0);
        v[t] = v1p;
        __syncthreads();
    }
    for (int it = 0; it < WW; ++it) {          // leftward fill
        const float dr = (t < WW - 1) ? d[t + 1] : 0.0f;
        const float vr = (t < WW - 1) ? v[t + 1] : 0.0f;
        const float dc = d[t], v0 = v[t];
        __syncthreads();
        const float v1  = v0 + vr;
        const float v1p = (v1 > 0.0f) ? 1.0f : 0.0f;
        d[t] = dc * v0 + (dc + dr) / (v1 + 1e-4f) * (v1p - v0);
        v[t] = v1p;
        __syncthreads();
    }
    out[(size_t)row * WW + t] = di * vmf + d[t] * (1.0f - vmf);
}

// ---------------------------------------------------------------------------
extern "C" void kernel_launch(void* const* d_in, const int* in_sizes, int n_in,
                              void* d_out, int out_size, void* d_ws, size_t ws_size,
                              hipStream_t stream)
{
    const float* cost_r2l = (const float*)d_in[0];
    const float* cost_l2r = (const float*)d_in[1];
    const float* params   = (const float*)d_in[2];
    const int*   mdp      = (const int*)d_in[3];
    float*       out      = (float*)d_out;

    const size_t VOL = (size_t)BB * CC * WW * KK;     // 1,572,864 floats
    float* S0      = (float*)d_ws;                    // softmax r2l
    float* S1      = S0 + VOL;                        // softmax l2r
    float* cm0     = S1 + VOL;                        // guidance r2l (B*W*K)
    float* cm1     = cm0 + (size_t)BB * WW * KK;
    float* dispsum = cm1 + (size_t)BB * WW * KK;      // B*C*W
    float* colsum  = dispsum + (size_t)BB * CC * WW;  // B*C*K

    // zero the atomic accumulators (dispsum & colsum are contiguous)
    (void)hipMemsetAsync(dispsum, 0, sizeof(float) * (size_t)BB * CC * (WW + KK), stream);

    band_softmax_kernel<<<dim3(BB * CC * WW, 2), 128, 0, stream>>>(
        cost_r2l, cost_l2r, mdp, S0, S1);

    cmean_kernel<<<dim3((WW * KK) / 256, 2, BB), 256, 0, stream>>>(S0, S1, cm0, cm1);

    bilateral_kernel<0><<<dim3(WW / 16, KK / 16, BB), dim3(16, 16), 0, stream>>>(
        S0, cm0, params, dispsum);
    bilateral_kernel<1><<<dim3(WW / 16, KK / 16, BB), dim3(16, 16), 0, stream>>>(
        S1, cm1, params, colsum);

    regress_fill_kernel<<<BB * CC, 128, 0, stream>>>(dispsum, colsum, out);
}
